// Rnn_69483980915479
// MI455X (gfx1250) — compile-verified, matrix operand order checked by reading, not run on hardware
//
#include <hip/hip_runtime.h>
#include <hip/hip_bf16.h>

typedef __attribute__((ext_vector_type(16))) _Float16 v16h;
typedef __attribute__((ext_vector_type(8)))  float    v8f;

#define B_  256
#define T_  512
#define H_  512
#define NG_ 2048   // 4*H
#define P_  128
#define KD_ 1024   // concat [xc | nh]

__device__ __forceinline__ float sigmoidf_(float v) { return 1.0f / (1.0f + __expf(-v)); }

union AFrag { uint4 q[2]; v16h v; };

// Wave-level GEMM: C(32x64) += A(32xK) * Bpacked(Kx64), WMMA f32_16x16x32_f16.
// A: row-major f16, lda=K. Bpacked: fragment-native tiles of 512 halves,
// tile index = ntile*(K/32)+ktile, lane l owns halves [l*16, l*16+16).
template<int K>
__device__ __forceinline__ void gemm_wave(const _Float16* __restrict__ A,
                                          const _Float16* __restrict__ Bp,
                                          int rowBase, int ntBase, int lane,
                                          v8f c[2][4]) {
  const int kTiles = K / 32;
  const int khi = (lane >> 4) * 8;      // lanes 16-31 start at K offset 8
  const int lrow = lane & 15;
  for (int kt = 0; kt < kTiles; ++kt) {
    AFrag a[2];
#pragma unroll
    for (int i = 0; i < 2; ++i) {
      const _Float16* ap = A + (size_t)(rowBase + i * 16 + lrow) * K + kt * 32 + khi;
      a[i].q[0] = *(const uint4*)ap;          // K offsets +0..7
      a[i].q[1] = *(const uint4*)(ap + 16);   // K offsets +16..23
    }
#pragma unroll
    for (int j = 0; j < 4; ++j) {
      const v16h bv = *(const v16h*)(Bp + ((size_t)(ntBase + j) * kTiles + kt) * 512 + lane * 16);
#pragma unroll
      for (int i = 0; i < 2; ++i) {
        c[i][j] = __builtin_amdgcn_wmma_f32_16x16x32_f16(
            false, a[i].v, false, bv, (short)0, c[i][j], false, false);
      }
    }
  }
}

// ---- weight packing: gate-interleaved (n = 4*unit + gate) + WMMA B-fragment layout ----
__device__ __forceinline__ void pack_coords(int idx, int kTiles, int& n, int& k) {
  int tile = idx >> 9;                 // 512 halves per 16x32 tile
  int within = idx & 511;
  int lane = within >> 4;
  int j = within & 15;                 // half index within v16h
  int nt = tile / kTiles, kt = tile % kTiles;
  n = nt * 16 + (lane & 15);
  int khi = lane >> 4;
  int vg = j >> 1, pos = j & 1;
  k = kt * 32 + (vg & 3) * 2 + pos + khi * 8 + (vg >> 2) * 16;
}

__global__ void pack_enc_kernel(const float* __restrict__ Whh, const float* __restrict__ Wih,
                                const float* __restrict__ bih, const float* __restrict__ bhh,
                                _Float16* __restrict__ WhhP, float* __restrict__ wihP,
                                float* __restrict__ biasP) {
  int idx = blockIdx.x * blockDim.x + threadIdx.x;
  if (idx < NG_ * H_) {
    int n, k; pack_coords(idx, H_ / 32, n, k);
    int unit = n >> 2, gate = n & 3;
    WhhP[idx] = (_Float16)Whh[(size_t)(gate * H_ + unit) * H_ + k];
  }
  if (idx < NG_) {
    int unit = idx >> 2, gate = idx & 3;
    int row = gate * H_ + unit;
    biasP[idx] = bih[row] + bhh[row];
    wihP[idx] = Wih[row];              // enc_Wih shape (4H,1)
  }
}

__global__ void pack_dec_kernel(const float* __restrict__ Wih, const float* __restrict__ Whh,
                                const float* __restrict__ bih, const float* __restrict__ bhh,
                                _Float16* __restrict__ WP, float* __restrict__ biasP) {
  int idx = blockIdx.x * blockDim.x + threadIdx.x;
  if (idx < NG_ * KD_) {
    int n, k; pack_coords(idx, KD_ / 32, n, k);
    int unit = n >> 2, gate = n & 3;
    int row = gate * H_ + unit;
    float v = (k < H_) ? Wih[(size_t)row * H_ + k] : Whh[(size_t)row * H_ + (k - H_)];
    WP[idx] = (_Float16)v;
  }
  if (idx < NG_) {
    int unit = idx >> 2, gate = idx & 3;
    int row = gate * H_ + unit;
    biasP[idx] = bih[row] + bhh[row];
  }
}

__global__ void init_enc_kernel(_Float16* h0, float* c0) {
  int idx = blockIdx.x * blockDim.x + threadIdx.x;
  if (idx < B_ * H_) { h0[idx] = (_Float16)0.0f; c0[idx] = 0.0f; }
}

__global__ void init_dec_kernel(const float* __restrict__ hF32, const float* __restrict__ cEnc,
                                _Float16* __restrict__ A0, float* __restrict__ xc0,
                                float* __restrict__ nc0) {
  int idx = blockIdx.x * blockDim.x + threadIdx.x;
  if (idx < B_ * H_) {
    int b = idx >> 9, u = idx & 511;
    float hv = hF32[idx];
    A0[(size_t)b * KD_ + u] = (_Float16)hv;          // xc half
    A0[(size_t)b * KD_ + H_ + u] = (_Float16)hv;     // nh half
    xc0[idx] = hv;
    nc0[idx] = cEnc[idx];
  }
}

// ---- one encoder timestep: gates = h@WhhP.T (WMMA) + x_t*wih + bias -> LSTM pointwise ----
__global__ __launch_bounds__(256) void enc_step_kernel(
    const _Float16* __restrict__ hPrev, const float* __restrict__ cPrev,
    const _Float16* __restrict__ WhhP, const float* __restrict__ wihP,
    const float* __restrict__ biasP, const float* __restrict__ x,
    _Float16* __restrict__ hNext, float* __restrict__ cNext,
    float* __restrict__ hF32, int t) {
  __shared__ float lds[8 * 32 * 64];   // 64 KB
  int tid = threadIdx.x, lane = tid & 31, wave = tid >> 5;
  int bM = blockIdx.x & 1, bN = blockIdx.x >> 1;
  int wM = wave & 3, wN = wave >> 2;
  int rowBase = bM * 128 + wM * 32;
  int nBase = bN * 128 + wN * 64;
  v8f c[2][4];
#pragma unroll
  for (int i = 0; i < 2; ++i)
#pragma unroll
    for (int j = 0; j < 4; ++j) c[i][j] = (v8f){0,0,0,0,0,0,0,0};

  gemm_wave<H_>(hPrev, WhhP, rowBase, nBase >> 4, lane, c);

  int waveBase = wave * 2048;
#pragma unroll
  for (int i = 0; i < 2; ++i)
#pragma unroll
    for (int j = 0; j < 4; ++j)
#pragma unroll
      for (int r = 0; r < 8; ++r) {
        int mL = i * 16 + ((lane >> 4) << 3) + r;
        int nL = j * 16 + (lane & 15);
        lds[waveBase + mL * 64 + nL] = c[i][j][r];
      }
  __syncthreads();

  for (int it = lane; it < 512; it += 32) {
    int mL = it >> 4, uL = it & 15;
    int row = rowBase + mL;
    int unit = (nBase >> 2) + uL;
    int n = unit * 4;
    const float* g4 = &lds[waveBase + mL * 64 + uL * 4];
    float xv = x[(size_t)row * T_ + t];
    float gi = g4[0] + xv * wihP[n + 0] + biasP[n + 0];
    float gf = g4[1] + xv * wihP[n + 1] + biasP[n + 1];
    float gg = g4[2] + xv * wihP[n + 2] + biasP[n + 2];
    float go = g4[3] + xv * wihP[n + 3] + biasP[n + 3];
    float cp = cPrev[(size_t)row * H_ + unit];
    float c2 = sigmoidf_(gf) * cp + sigmoidf_(gi) * tanhf(gg);
    float h2 = sigmoidf_(go) * tanhf(c2);
    cNext[(size_t)row * H_ + unit] = c2;
    hNext[(size_t)row * H_ + unit] = (_Float16)h2;
    hF32[(size_t)row * H_ + unit] = h2;
  }
}

// ---- Genc = h_enc @ enc_WhhP.T + biasEncP (constant across decoder steps) ----
__global__ __launch_bounds__(256) void genc_kernel(
    const _Float16* __restrict__ hEnc, const _Float16* __restrict__ WhhP,
    const float* __restrict__ biasP, float* __restrict__ Genc) {
  int tid = threadIdx.x, lane = tid & 31, wave = tid >> 5;
  int bM = blockIdx.x & 1, bN = blockIdx.x >> 1;
  int wM = wave & 3, wN = wave >> 2;
  int rowBase = bM * 128 + wM * 32;
  int nBase = bN * 128 + wN * 64;
  v8f c[2][4];
#pragma unroll
  for (int i = 0; i < 2; ++i)
#pragma unroll
    for (int j = 0; j < 4; ++j) c[i][j] = (v8f){0,0,0,0,0,0,0,0};

  gemm_wave<H_>(hEnc, WhhP, rowBase, nBase >> 4, lane, c);

#pragma unroll
  for (int i = 0; i < 2; ++i)
#pragma unroll
    for (int j = 0; j < 4; ++j)
#pragma unroll
      for (int r = 0; r < 8; ++r) {
        int row = rowBase + i * 16 + ((lane >> 4) << 3) + r;
        int n = nBase + j * 16 + (lane & 15);
        Genc[(size_t)row * NG_ + n] = c[i][j][r] + biasP[n];
      }
}

// ---- one decoder timestep ----
// blocks 0..31: dec-LSTM GEMM (K=1024 over [xc|nh]) + pointwise -> nh',nc'
// blocks 32..39: y = xc . dense_W + b  ->  x_next = lstm(y, h_enc, c_enc) via Genc
__global__ __launch_bounds__(256) void dec_step_kernel(
    const _Float16* __restrict__ Acur, const float* __restrict__ ncPrev,
    const float* __restrict__ xc32cur,
    const _Float16* __restrict__ decWP, const float* __restrict__ biasDecP,
    const float* __restrict__ Genc, const float* __restrict__ cEnc,
    const float* __restrict__ wihEncP, const float* __restrict__ denseW,
    const float* __restrict__ denseB,
    _Float16* __restrict__ Anext, float* __restrict__ ncNext,
    float* __restrict__ xc32next, float* __restrict__ out, int t) {
  __shared__ float lds[8 * 32 * 64];
  int bid = blockIdx.x;
  int tid = threadIdx.x, lane = tid & 31, wave = tid >> 5;

  if (bid < 32) {
    int bM = bid & 1, bN = bid >> 1;
    int wM = wave & 3, wN = wave >> 2;
    int rowBase = bM * 128 + wM * 32;
    int nBase = bN * 128 + wN * 64;
    v8f c[2][4];
#pragma unroll
    for (int i = 0; i < 2; ++i)
#pragma unroll
      for (int j = 0; j < 4; ++j) c[i][j] = (v8f){0,0,0,0,0,0,0,0};

    gemm_wave<KD_>(Acur, decWP, rowBase, nBase >> 4, lane, c);

    int waveBase = wave * 2048;
#pragma unroll
    for (int i = 0; i < 2; ++i)
#pragma unroll
      for (int j = 0; j < 4; ++j)
#pragma unroll
        for (int r = 0; r < 8; ++r) {
          int mL = i * 16 + ((lane >> 4) << 3) + r;
          int nL = j * 16 + (lane & 15);
          lds[waveBase + mL * 64 + nL] = c[i][j][r];
        }
    __syncthreads();

    for (int it = lane; it < 512; it += 32) {
      int mL = it >> 4, uL = it & 15;
      int row = rowBase + mL;
      int unit = (nBase >> 2) + uL;
      int n = unit * 4;
      const float* g4 = &lds[waveBase + mL * 64 + uL * 4];
      float gi = g4[0] + biasDecP[n + 0];
      float gf = g4[1] + biasDecP[n + 1];
      float gg = g4[2] + biasDecP[n + 2];
      float go = g4[3] + biasDecP[n + 3];
      float cp = ncPrev[(size_t)row * H_ + unit];
      float c2 = sigmoidf_(gf) * cp + sigmoidf_(gi) * tanhf(gg);
      float h2 = sigmoidf_(go) * tanhf(c2);
      ncNext[(size_t)row * H_ + unit] = c2;
      Anext[(size_t)row * KD_ + H_ + unit] = (_Float16)h2;   // nh half for next step
    }
  } else {
    int b0 = (bid - 32) * 32;
    float* red = lds;          // 32 x 8 partials
    float* yv = lds + 256;     // 32 y values
    int r = tid >> 3, seg = tid & 7;
    {
      const float* xr = xc32cur + (size_t)(b0 + r) * H_;
      float s = 0.0f;
      for (int k = 0; k < 64; ++k) s += xr[seg * 64 + k] * denseW[seg * 64 + k];
      red[r * 8 + seg] = s;
    }
    __syncthreads();
    if (tid < 32) {
      float y = denseB[0];
      for (int s = 0; s < 8; ++s) y += red[tid * 8 + s];
      yv[tid] = y;
      out[(size_t)(b0 + tid) * P_ + t] = y;
    }
    __syncthreads();
    for (int idx = tid; idx < 32 * 512; idx += 256) {
      int m = idx >> 9, u = idx & 511;
      int b = b0 + m;
      float y = yv[m];
      size_t gb = (size_t)b * NG_ + (size_t)u * 4;
      float gi = y * wihEncP[u * 4 + 0] + Genc[gb + 0];
      float gf = y * wihEncP[u * 4 + 1] + Genc[gb + 1];
      float gg = y * wihEncP[u * 4 + 2] + Genc[gb + 2];
      float go = y * wihEncP[u * 4 + 3] + Genc[gb + 3];
      float cp = cEnc[(size_t)b * H_ + u];
      float c2 = sigmoidf_(gf) * cp + sigmoidf_(gi) * tanhf(gg);
      float x2 = sigmoidf_(go) * tanhf(c2);
      Anext[(size_t)b * KD_ + u] = (_Float16)x2;             // xc half for next step
      xc32next[(size_t)b * H_ + u] = x2;
    }
  }
}

extern "C" void kernel_launch(void* const* d_in, const int* in_sizes, int n_in,
                              void* d_out, int out_size, void* d_ws, size_t ws_size,
                              hipStream_t stream) {
  (void)in_sizes; (void)n_in; (void)out_size; (void)ws_size;
  const float* x      = (const float*)d_in[0];
  const float* encWih = (const float*)d_in[1];
  const float* encWhh = (const float*)d_in[2];
  const float* encBih = (const float*)d_in[3];
  const float* encBhh = (const float*)d_in[4];
  const float* decWih = (const float*)d_in[5];
  const float* decWhh = (const float*)d_in[6];
  const float* decBih = (const float*)d_in[7];
  const float* decBhh = (const float*)d_in[8];
  const float* denseW = (const float*)d_in[9];
  const float* denseB = (const float*)d_in[10];
  float* out = (float*)d_out;

  char* ws = (char*)d_ws;
  size_t off = 0;
  auto alloc = [&](size_t bytes) -> void* {
    void* p = ws + off;
    off = (off + bytes + 255) & ~(size_t)255;
    return p;
  };
  _Float16* WhhP   = (_Float16*)alloc((size_t)NG_ * H_ * 2);
  _Float16* decWP  = (_Float16*)alloc((size_t)NG_ * KD_ * 2);
  float* biasEncP  = (float*)alloc(NG_ * 4);
  float* biasDecP  = (float*)alloc(NG_ * 4);
  float* wihEncP   = (float*)alloc(NG_ * 4);
  _Float16* hH[2]  = {(_Float16*)alloc((size_t)B_ * H_ * 2), (_Float16*)alloc((size_t)B_ * H_ * 2)};
  float* cF[2]     = {(float*)alloc((size_t)B_ * H_ * 4), (float*)alloc((size_t)B_ * H_ * 4)};
  float* hF32      = (float*)alloc((size_t)B_ * H_ * 4);
  float* Genc      = (float*)alloc((size_t)B_ * NG_ * 4);
  _Float16* Adec[2] = {(_Float16*)alloc((size_t)B_ * KD_ * 2), (_Float16*)alloc((size_t)B_ * KD_ * 2)};
  float* xc32[2]   = {(float*)alloc((size_t)B_ * H_ * 4), (float*)alloc((size_t)B_ * H_ * 4)};
  float* nc[2]     = {(float*)alloc((size_t)B_ * H_ * 4), (float*)alloc((size_t)B_ * H_ * 4)};

  pack_enc_kernel<<<(NG_ * H_ + 255) / 256, 256, 0, stream>>>(encWhh, encWih, encBih, encBhh,
                                                              WhhP, wihEncP, biasEncP);
  pack_dec_kernel<<<(NG_ * KD_ + 255) / 256, 256, 0, stream>>>(decWih, decWhh, decBih, decBhh,
                                                               decWP, biasDecP);
  init_enc_kernel<<<(B_ * H_ + 255) / 256, 256, 0, stream>>>(hH[0], cF[0]);

  for (int t = 0; t < T_; ++t) {
    int cur = t & 1, nxt = cur ^ 1;
    enc_step_kernel<<<32, 256, 0, stream>>>(hH[cur], cF[cur], WhhP, wihEncP, biasEncP, x,
                                            hH[nxt], cF[nxt], hF32, t);
  }
  // T_ is even -> final encoder state sits in index 0
  _Float16* hEncH = hH[0];
  float* cEnc = cF[0];

  genc_kernel<<<32, 256, 0, stream>>>(hEncH, WhhP, biasEncP, Genc);
  init_dec_kernel<<<(B_ * H_ + 255) / 256, 256, 0, stream>>>(hF32, cEnc, Adec[0], xc32[0], nc[0]);

  for (int t = 0; t < P_; ++t) {
    int cur = t & 1, nxt = cur ^ 1;
    dec_step_kernel<<<40, 256, 0, stream>>>(Adec[cur], nc[cur], xc32[cur], decWP, biasDecP,
                                            Genc, cEnc, wihEncP, denseW, denseB,
                                            Adec[nxt], nc[nxt], xc32[nxt], out, t);
  }
}
